// DeformableAttention1D_71176198029953
// MI455X (gfx1250) — matrix operand order repaired and checked by reading, not compiled, not run on hardware
//
#include <hip/hip_runtime.h>

// ---------------------------------------------------------------------------
// DeformableAttention1D forward for MI455X (gfx1250, wave32, WMMA).
// All heavy matrix math runs on v_wmma_f32_16x16x32_bf16.
// ---------------------------------------------------------------------------

typedef __bf16 bf16_t;
typedef __attribute__((ext_vector_type(16))) __bf16 v16bf;
typedef __attribute__((ext_vector_type(8)))  __bf16 v8bf;
typedef __attribute__((ext_vector_type(8)))  float  v8f;

// ---------------- workspace layout (bytes) ----------------
static constexpr size_t OFF_XBF   = 0;                                  // x bf16          [2048][512]
static constexpr size_t OFF_WQT   = OFF_XBF   + 2048ull*512*2;          // w_q^T bf16      [512][512]
static constexpr size_t OFF_WKVT  = OFF_WQT   + 512ull*512*2;           // w_kv^T bf16     [1024][512]
static constexpr size_t OFF_WOUTT = OFF_WKVT  + 1024ull*512*2;          // w_out^T bf16    [512][512]
static constexpr size_t OFF_W2T   = OFF_WOUTT + 512ull*512*2;           // cpb_w2^T bf16   [128][128]
static constexpr size_t OFF_QF    = OFF_W2T   + 128ull*128*2;           // q f32           [2048][512]
static constexpr size_t OFF_QH    = OFF_QF    + 2048ull*512*4;          // q*scale bf16    [b][h][n][dh]
static constexpr size_t OFF_VGS   = OFF_QH    + 2ull*8*1024*64*2;       // vgrid_scaled f32 [4][256]
static constexpr size_t OFF_KVFBF = OFF_VGS   + 4096;                   // kv_feats bf16   [b][m][512]
static constexpr size_t OFF_KVF   = OFF_KVFBF + 2ull*256*512*2;         // kv f32          [b*m][1024]
static constexpr size_t OFF_KH    = OFF_KVF   + 2ull*256*1024*4;        // k bf16          [b][h][m][dh]
static constexpr size_t OFF_VHT   = OFF_KH    + 2ull*8*256*64*2;        // v^T bf16        [b][h][dh][m]
static constexpr size_t OFF_BIAS  = OFF_VHT   + 2ull*8*64*256*2;        // cpb bias f32    [b][h][n][m]
static constexpr size_t OFF_AOUT  = OFF_BIAS  + 2ull*8*1024*256*4;      // attn out bf16   [b][n][512]
// total ~= 33.1 MB

// ---------------- WMMA fragment helpers (ISA 7.12.2 layouts) ----------------
// A 16x32 bf16: lane L holds row M=L&15; K = (e<8?0:16) + ((L>=16)?8:0) + (e&7)
__device__ __forceinline__ v16bf load_a_frag(const bf16_t* __restrict__ base, int lane, int ld) {
  const int m  = lane & 15;
  const int kh = (lane >> 4) & 1;
  const bf16_t* p = base + (size_t)m * ld + kh * 8;
  v8bf lo = *(const v8bf*)(p);        // K = kh*8 + 0..7
  v8bf hi = *(const v8bf*)(p + 16);   // K = 16 + kh*8 + 0..7
  v16bf a;
#pragma unroll
  for (int i = 0; i < 8; ++i) { a[i] = lo[i]; a[i + 8] = hi[i]; }
  return a;
}

// B 32x16 bf16 from an [N][K]-transposed matrix: lane L holds col N=L&15;
// K = (e<8?0:16) + ((L>=16)?8:0) + (e&7)   (same K order as A)
__device__ __forceinline__ v16bf load_b_frag(const bf16_t* __restrict__ base, int lane, int ld) {
  const int n  = lane & 15;
  const int kh = (lane >> 4) & 1;
  const bf16_t* p = base + (size_t)n * ld + kh * 8;
  v8bf lo = *(const v8bf*)(p);
  v8bf hi = *(const v8bf*)(p + 16);
  v16bf b;
#pragma unroll
  for (int i = 0; i < 8; ++i) { b[i] = lo[i]; b[i + 8] = hi[i]; }
  return b;
}

__device__ __forceinline__ v8f wmma_bf16(v16bf a, v16bf b, v8f c) {
  return __builtin_amdgcn_wmma_f32_16x16x32_bf16(false, a, false, b, (short)0, c, false, false);
}
// C/D f32 16x16: VGPR r, lanes 0-15 -> (row=r, col=lane); lanes 16-31 -> (row=8+r, col=lane-16)

// ---------------- conversion kernels ----------------
__global__ void k_cvt_bf16(const float* __restrict__ src, bf16_t* __restrict__ dst, int n) {
  int i = blockIdx.x * 256 + threadIdx.x;
  if (i < n) dst[i] = (bf16_t)src[i];
}

// dst[c*R + r] = bf16(src[r*C + c])  (produce [N][K] weight layouts)
__global__ void k_transpose_bf16(const float* __restrict__ src, bf16_t* __restrict__ dst,
                                 int R, int C) {
  int i = blockIdx.x * 256 + threadIdx.x;
  if (i < R * C) {
    int r = i / C, c = i - r * C;
    dst[(size_t)c * R + r] = (bf16_t)src[i];
  }
}

// ---------------- generic WMMA GEMM: C[M][N] = A[M][K] * Bt[N][K]^T (+bias) ----------------
// block = 256 threads = 8 waves; wave -> 16 rows x 64 cols (4 accs share one A frag);
// software-pipelined K loop so next-iteration loads overlap current WMMAs.
// grid = (M/128, N/64)
__global__ void k_gemm_bf16(const bf16_t* __restrict__ A, const bf16_t* __restrict__ Bt,
                            float* __restrict__ C, const float* __restrict__ bias,
                            int M, int N, int K) {
  const int lane = threadIdx.x & 31;
  const int wave = threadIdx.x >> 5;
  const int row0 = (blockIdx.x * 8 + wave) * 16;
  const int col0 = blockIdx.y * 64;
  if (row0 >= M || col0 >= N) return;
  v8f acc[4] = {};
  const bf16_t* arow = A + (size_t)row0 * K;
  const bf16_t* bp[4];
#pragma unroll
  for (int t = 0; t < 4; ++t) bp[t] = Bt + (size_t)(col0 + t * 16) * K;

  // prologue: fragments for k = 0
  v16bf a = load_a_frag(arow, lane, K);
  v16bf bb[4];
#pragma unroll
  for (int t = 0; t < 4; ++t) bb[t] = load_b_frag(bp[t], lane, K);

  for (int k = 32; k < K; k += 32) {
    // issue next-iteration loads first so they overlap this iteration's WMMAs
    __builtin_prefetch(arow + k, 0, 1);
    v16bf an = load_a_frag(arow + k, lane, K);
    v16bf bn[4];
#pragma unroll
    for (int t = 0; t < 4; ++t) bn[t] = load_b_frag(bp[t] + k, lane, K);
#pragma unroll
    for (int t = 0; t < 4; ++t) acc[t] = wmma_bf16(a, bb[t], acc[t]);
    a = an;
#pragma unroll
    for (int t = 0; t < 4; ++t) bb[t] = bn[t];
  }
#pragma unroll
  for (int t = 0; t < 4; ++t) acc[t] = wmma_bf16(a, bb[t], acc[t]);

  const int rowo = (lane >> 4) * 8;
  const int col  = lane & 15;
#pragma unroll
  for (int t = 0; t < 4; ++t) {
    const int c0 = col0 + t * 16 + col;
    const float bb0 = bias ? bias[c0] : 0.f;
#pragma unroll
    for (int r = 0; r < 8; ++r) {
      const int row = row0 + rowo + r;
      C[(size_t)row * N + c0] = acc[t][r] + bb0;
    }
  }
}

// ---------------- offset net + collapsed grid-sample -> kv_feats ----------------
// grid = 4 blocks (bg), 256 threads. Tiny (<0.1% of FLOPs): scalar VALU is fine.
__global__ void k_offsets(const float* __restrict__ q, const float* __restrict__ c1w,
                          const float* __restrict__ c1b, const float* __restrict__ c2w,
                          float* __restrict__ vgs, bf16_t* __restrict__ kvf) {
  const int bg = blockIdx.x, b = bg >> 1, g = bg & 1;
  const int t = threadIdx.x;
  __shared__ float rowv[256];
  // grid_sample row: 0.5*(gf[:,:,511] + gf[:,:,512]); gf[bg,d,n] = q[b,n,g*256+d]
  {
    float a = q[((size_t)b * 1024 + 511) * 512 + g * 256 + t];
    float c = q[((size_t)b * 1024 + 512) * 512 + g * 256 + t];
    rowv[t] = 0.5f * (a + c);
  }
  __syncthreads();
  const int m = t;
  float acc = 0.f;
  for (int d = 0; d < 256; ++d) {
    float hv = c1b[d];
#pragma unroll
    for (int kk = 0; kk < 6; ++kk) {          // depthwise conv, stride 4, pad 1
      int n = m * 4 - 1 + kk;
      float v = (n >= 0 && n < 1024) ? q[((size_t)b * 1024 + n) * 512 + g * 256 + d] : 0.f;
      hv += v * c1w[d * 6 + kk];
    }
    float ge = 0.5f * hv * (1.f + erff(hv * 0.70710678118654752f)); // exact GELU
    acc += ge * c2w[d];                        // 1x1 conv reduce
  }
  float off = tanhf(acc) * 4.0f;
  float vg  = 2.0f * ((float)m + off) * (1.0f / 255.0f) - 1.0f;
  vgs[bg * 256 + m] = vg;
  // grid_sample over W=1 collapses to a single column weight
  float ix  = vg * 0.5f;
  float x0  = floorf(ix);
  float wx1 = ix - x0;
  float w = (x0 == 0.0f ? (1.0f - wx1) : 0.0f) + (x0 == -1.0f ? wx1 : 0.0f);
  for (int d = 0; d < 256; ++d)                // kv_feats[b][m][g*256+d] = row[d]*w
    kvf[((size_t)b * 256 + m) * 512 + g * 256 + d] = (bf16_t)(rowv[d] * w);
}

// ---------------- pack q/k/v into head-major bf16 layouts ----------------
__global__ void k_pack(const float* __restrict__ qf, const float* __restrict__ kvf32,
                       bf16_t* __restrict__ qh, bf16_t* __restrict__ kh,
                       bf16_t* __restrict__ vht) {
  const int idx = blockIdx.x * 256 + threadIdx.x;      // grid covers 2^20
  {
    const int dh = idx & 63, n = (idx >> 6) & 1023, h = (idx >> 16) & 7, b = idx >> 19;
    qh[idx] = (bf16_t)(qf[((size_t)b * 1024 + n) * 512 + h * 64 + dh] * 0.125f); // 64^-0.5
  }
  if (idx < 2 * 8 * 256 * 64) {
    const int dh = idx & 63, m = (idx >> 6) & 255, h = (idx >> 14) & 7, b = idx >> 17;
    const size_t base = ((size_t)b * 256 + m) * 1024 + h * 64 + dh;
    kh[idx] = (bf16_t)kvf32[base];                                   // [b][h][m][dh]
    vht[(((size_t)b * 8 + h) * 64 + dh) * 256 + m] = (bf16_t)kvf32[base + 512]; // [b][h][dh][m]
  }
}

// ---------------- CPB relative-position bias MLP (dominant compute) ----------------
// 1M rows of (1 -> 128 -> 128 -> 4). Block = 128 rows (fixed bg,n; half of m), 8 waves.
// Layer2 (128x128) runs on WMMA with the wave's 4 A-fragments hoisted into registers
// (read from LDS once, reused across all 8 N-tiles); layer3 folded into per-lane FMAs.
__global__ void k_cpb(const float* __restrict__ vgs, const float* __restrict__ w1,
                      const float* __restrict__ b1, const bf16_t* __restrict__ w2t,
                      const float* __restrict__ b2, const float* __restrict__ w3,
                      const float* __restrict__ b3, float* __restrict__ biasOut) {
  const int bid = blockIdx.x;                 // 8192 blocks
  const int mhalf = bid & 1;
  const int n  = (bid >> 1) & 1023;
  const int bg = bid >> 11;
  const int m0 = mhalf * 128;
  const int t  = threadIdx.x;
  __shared__ float poss[128];
  __shared__ __align__(16) bf16_t h1[128][128];   // 32 KB
  const float seq = 2.0f * (float)n * (1.0f / 1023.0f) - 1.0f;
  if (t < 128) {
    float p = seq - vgs[bg * 256 + m0 + t];
    float s = (p > 0.f) ? 1.f : ((p < 0.f) ? -1.f : 0.f);
    poss[t] = s * log1pf(fabsf(p));
  }
  __syncthreads();
  for (int i = t; i < 128 * 128; i += 256) {       // layer1: scalar-in, relu
    int r = i >> 7, c = i & 127;
    float v = poss[r] * w1[c] + b1[c];
    h1[r][c] = (bf16_t)(v > 0.f ? v : 0.f);
  }
  __syncthreads();
  const int lane = t & 31, wave = t >> 5;
  const int r0   = wave * 16;
  const int rowo = (lane >> 4) * 8;
  const int col  = lane & 15;
  // hoist the wave's A fragments (16x128 of h1) out of the N-tile loop
  v16bf afr[4];
#pragma unroll
  for (int kc = 0; kc < 4; ++kc) afr[kc] = load_a_frag(&h1[r0][kc * 32], lane, 128);
  float p3[8][4];
#pragma unroll
  for (int r = 0; r < 8; ++r)
#pragma unroll
    for (int o = 0; o < 4; ++o) p3[r][o] = 0.f;
  for (int nt = 0; nt < 8; ++nt) {                 // layer2: 8 N-tiles x 4 K-chunks
    v8f acc = {};
#pragma unroll
    for (int kc = 0; kc < 4; ++kc) {
      v16bf bw = load_b_frag(w2t + (size_t)(nt * 16) * 128 + kc * 32, lane, 128);
      acc = wmma_bf16(afr[kc], bw, acc);
    }
    const int c2 = nt * 16 + col;
    const float bb = b2[c2];
#pragma unroll
    for (int r = 0; r < 8; ++r) {                  // fused relu + layer3 partials
      float hv = acc[r] + bb; hv = hv > 0.f ? hv : 0.f;
#pragma unroll
      for (int o = 0; o < 4; ++o) p3[r][o] += hv * w3[c2 * 4 + o];
    }
  }
#pragma unroll
  for (int r = 0; r < 8; ++r)                      // reduce over the 16 col-lanes
#pragma unroll
    for (int o = 0; o < 4; ++o) {
      float v = p3[r][o];
      for (int sh = 1; sh < 16; sh <<= 1) v += __shfl_xor(v, sh, 32);
      p3[r][o] = v;
    }
  if (col == 0) {
    const int b = bg >> 1, g = bg & 1;
#pragma unroll
    for (int r = 0; r < 8; ++r) {
      const int m = m0 + r0 + rowo + r;
#pragma unroll
      for (int o = 0; o < 4; ++o)   // head = g*4 + o
        biasOut[(((size_t)(b * 8 + g * 4 + o) * 1024 + n) * 256) + m] = p3[r][o] + b3[o];
    }
  }
}

// ---------------- flash-style attention: softmax(QK^T + bias) @ V ----------------
// grid = (8, heads=8, b=2); block = 256 thr = 8 waves; wave = 16 query rows, streams m in 32s.
__global__ void k_attn(const bf16_t* __restrict__ qh, const bf16_t* __restrict__ kh,
                       const bf16_t* __restrict__ vht, const float* __restrict__ biasT,
                       bf16_t* __restrict__ aout) {
  const int b = blockIdx.z, h = blockIdx.y;
  const int lane = threadIdx.x & 31, wave = threadIdx.x >> 5;
  const int n0 = blockIdx.x * 128 + wave * 16;
  const bf16_t* qbase = qh  + ((size_t)(b * 8 + h) * 1024 + n0) * 64;
  const bf16_t* kbase = kh  + (size_t)(b * 8 + h) * 256 * 64;
  const bf16_t* vbase = vht + (size_t)(b * 8 + h) * 64 * 256;
  const float*  bbase = biasT + ((size_t)(b * 8 + h) * 1024 + n0) * 256;
  __shared__ __align__(16) bf16_t pl[8][16][32];   // probs re-layout tile, 1 KB/wave
  v16bf aq0 = load_a_frag(qbase, lane, 64);        // Q 16x64 kept in registers
  v16bf aq1 = load_a_frag(qbase + 32, lane, 64);
  v8f accO[4] = {};
  float mrow[8], lrow[8];
#pragma unroll
  for (int r = 0; r < 8; ++r) { mrow[r] = -1e30f; lrow[r] = 0.f; }
  const int rowo = (lane >> 4) * 8;
  const int col  = lane & 15;
  for (int mc = 0; mc < 256; mc += 32) {
    v8f s0 = {}, s1 = {};
    {
      v16bf bk0 = load_b_frag(kbase + (size_t)mc * 64, lane, 64);
      v16bf bk1 = load_b_frag(kbase + (size_t)mc * 64 + 32, lane, 64);
      s0 = wmma_bf16(aq0, bk0, s0);
      s0 = wmma_bf16(aq1, bk1, s0);
    }
    {
      v16bf bk0 = load_b_frag(kbase + (size_t)(mc + 16) * 64, lane, 64);
      v16bf bk1 = load_b_frag(kbase + (size_t)(mc + 16) * 64 + 32, lane, 64);
      s1 = wmma_bf16(aq0, bk0, s1);
      s1 = wmma_bf16(aq1, bk1, s1);
    }
#pragma unroll
    for (int r = 0; r < 8; ++r) {                  // online softmax over 32 keys
      const float* brow = bbase + (size_t)(rowo + r) * 256 + mc;
      float sv0 = s0[r] + brow[col];
      float sv1 = s1[r] + brow[16 + col];
      float v = fmaxf(sv0, sv1);
      for (int sh = 1; sh < 16; sh <<= 1) v = fmaxf(v, __shfl_xor(v, sh, 32));
      const float nm    = fmaxf(mrow[r], v);
      const float scale = __expf(mrow[r] - nm);
      const float p0 = __expf(sv0 - nm);
      const float p1 = __expf(sv1 - nm);
      float ps = p0 + p1;
      for (int sh = 1; sh < 16; sh <<= 1) ps += __shfl_xor(ps, sh, 32);
      lrow[r] = lrow[r] * scale + ps;
      mrow[r] = nm;
#pragma unroll
      for (int t4 = 0; t4 < 4; ++t4) accO[t4][r] *= scale;
      pl[wave][rowo + r][col]      = (bf16_t)p0;   // C-layout -> A-layout via LDS
      pl[wave][rowo + r][16 + col] = (bf16_t)p1;
    }
    __syncthreads();
    v16bf ap = load_a_frag(&pl[wave][0][0], lane, 32);
#pragma unroll
    for (int t4 = 0; t4 < 4; ++t4) {               // P(16x32) @ V(32x64)
      v16bf bv = load_b_frag(vbase + (size_t)(t4 * 16) * 256 + mc, lane, 256);
      accO[t4] = wmma_bf16(ap, bv, accO[t4]);
    }
    __syncthreads();
  }
#pragma unroll
  for (int t4 = 0; t4 < 4; ++t4)
#pragma unroll
    for (int r = 0; r < 8; ++r) {
      const int n  = n0 + rowo + r;
      const int dh = t4 * 16 + col;
      aout[((size_t)b * 1024 + n) * 512 + h * 64 + dh] = (bf16_t)(accO[t4][r] / lrow[r]);
    }
}

// ---------------- host-side launch ----------------
extern "C" void kernel_launch(void* const* d_in, const int* in_sizes, int n_in,
                              void* d_out, int out_size, void* d_ws, size_t ws_size,
                              hipStream_t stream) {
  (void)in_sizes; (void)n_in; (void)out_size; (void)ws_size;
  const float* x      = (const float*)d_in[0];
  const float* w_q    = (const float*)d_in[1];
  const float* conv1w = (const float*)d_in[2];
  const float* conv1b = (const float*)d_in[3];
  const float* conv2w = (const float*)d_in[4];
  const float* cpbw1  = (const float*)d_in[5];
  const float* cpbb1  = (const float*)d_in[6];
  const float* cpbw2  = (const float*)d_in[7];
  const float* cpbb2  = (const float*)d_in[8];
  const float* cpbw3  = (const float*)d_in[9];
  const float* cpbb3  = (const float*)d_in[10];
  const float* w_kv   = (const float*)d_in[11];
  const float* w_out  = (const float*)d_in[12];
  const float* b_out  = (const float*)d_in[13];

  char* ws = (char*)d_ws;
  bf16_t* xbf   = (bf16_t*)(ws + OFF_XBF);
  bf16_t* wqT   = (bf16_t*)(ws + OFF_WQT);
  bf16_t* wkvT  = (bf16_t*)(ws + OFF_WKVT);
  bf16_t* woutT = (bf16_t*)(ws + OFF_WOUTT);
  bf16_t* w2T   = (bf16_t*)(ws + OFF_W2T);
  float*  qf    = (float*)(ws + OFF_QF);
  bf16_t* qh    = (bf16_t*)(ws + OFF_QH);
  float*  vgs   = (float*)(ws + OFF_VGS);
  bf16_t* kvfbf = (bf16_t*)(ws + OFF_KVFBF);
  float*  kvf   = (float*)(ws + OFF_KVF);
  bf16_t* kh    = (bf16_t*)(ws + OFF_KH);
  bf16_t* vht   = (bf16_t*)(ws + OFF_VHT);
  float*  biasb = (float*)(ws + OFF_BIAS);
  bf16_t* aoutb = (bf16_t*)(ws + OFF_AOUT);

  // 1) one-time conversions / weight transposes
  k_cvt_bf16<<<4096, 256, 0, stream>>>(x, xbf, 2048 * 512);
  k_transpose_bf16<<<1024, 256, 0, stream>>>(w_q,   wqT,   512, 512);
  k_transpose_bf16<<<2048, 256, 0, stream>>>(w_kv,  wkvT,  512, 1024);
  k_transpose_bf16<<<1024, 256, 0, stream>>>(w_out, woutT, 512, 512);
  k_transpose_bf16<<<64,   256, 0, stream>>>(cpbw2, w2T,   128, 128);

  // 2) q = x @ w_q  (2048x512x512)
  { dim3 g(16, 8); k_gemm_bf16<<<g, 256, 0, stream>>>(xbf, wqT, qf, nullptr, 2048, 512, 512); }

  // 3) offset net + collapsed grid-sample -> kv_feats (bf16) and vgrid_scaled
  k_offsets<<<4, 256, 0, stream>>>(qf, conv1w, conv1b, conv2w, vgs, kvfbf);

  // 4) kv = kv_feats @ w_kv  (512x1024x512)
  { dim3 g(4, 16); k_gemm_bf16<<<g, 256, 0, stream>>>(kvfbf, wkvT, kvf, nullptr, 512, 1024, 512); }

  // 5) pack q (scaled), k, v^T into head-major bf16 layouts
  k_pack<<<4096, 256, 0, stream>>>(qf, kvf, qh, kh, vht);

  // 6) CPB bias MLP (the dominant 34 GFLOP) -> bias[b][h][n][m]
  k_cpb<<<8192, 256, 0, stream>>>(vgs, cpbw1, cpbb1, w2T, cpbb2, cpbw3, cpbb3, biasb);

  // 7) fused attention: softmax(QK^T + bias) @ V
  { dim3 g(8, 8, 2); k_attn<<<g, 256, 0, stream>>>(qh, kh, vht, biasb, aoutb); }

  // 8) out = attn_out @ w_out + b_out  (2048x512x512) -> d_out (fp32)
  { dim3 g(16, 8); k_gemm_bf16<<<g, 256, 0, stream>>>(aoutb, woutT, (float*)d_out, b_out, 2048, 512, 512); }
}
